// MixBlock_85272280695451
// MI455X (gfx1250) — matrix operand order repaired and checked by reading, not compiled, not run on hardware
//
#include <hip/hip_runtime.h>
#include <math.h>
#include <stdint.h>

typedef _Float16 f16;
typedef __attribute__((ext_vector_type(16))) _Float16 v16h;
typedef __attribute__((ext_vector_type(8)))  float    v8f;

#define DIMC   384
#define DSAC   192
#define DCONVC 192
#define HEADS  8
#define DHEAD  24
#define HIDC   1536
#define BATCH  4
#define HH     56
#define WWD    56
#define HWC    (HH*WWD)    /* 3136 */
#define HDS    28
#define WDS    28
#define N2C    (HDS*WDS)   /* 784  */
#define EPSB   1e-5f
#define ATT_SCALE 0.2041241452319315f /* 24^-0.5 */
#define LDT    36          /* padded LDS row stride in dwords (144B, 16B aligned) */

// ---------------------------------------------------------------- WMMA helpers
__device__ __forceinline__ v8f wmma16(v16h a, v16h b, v8f c) {
  return __builtin_amdgcn_wmma_f32_16x16x32_f16(false, a, false, b, (short)0, c,
                                                false, false);
}
__device__ __forceinline__ float redmax16(float v) {
#pragma unroll
  for (int m = 1; m <= 8; m <<= 1) v = fmaxf(v, __shfl_xor(v, m, 32));
  return v;
}
__device__ __forceinline__ float redsum16(float v) {
#pragma unroll
  for (int m = 1; m <= 8; m <<= 1) v += __shfl_xor(v, m, 32);
  return v;
}
__device__ __forceinline__ float redsum32(float v) {
#pragma unroll
  for (int m = 1; m <= 16; m <<= 1) v += __shfl_xor(v, m, 32);
  return v;
}

// ------------------------------------------------- CDNA5 async global->LDS DMA
__device__ __forceinline__ void async_load16(unsigned lds_off, const void* g) {
  asm volatile("global_load_async_to_lds_b128 %0, %1, off"
               :
               : "v"(lds_off), "v"((unsigned long long)(size_t)g)
               : "memory");
}
__device__ __forceinline__ void wait_async0() {
  asm volatile("s_wait_asynccnt 0x0" ::: "memory");
}

// ---------------------------------------------------------------- tiled GEMM
// Out[M,N] = A[M,K] (row major f32) x W[N,K]^T + bias.
// 4 waves/WG, 64x64 WG tile, K-step 32, async double-buffered LDS staging,
// f16 WMMA fragments with f32 accumulation. Requires M%64==0, N%64==0, K%32==0.
__global__ __launch_bounds__(128) void k_gemm(
    const float* __restrict__ A, const float* __restrict__ W,
    const float* __restrict__ bias, float* __restrict__ Out,
    int M, int N, int K) {
  __shared__ float ldsA[2][64 * LDT];
  __shared__ float ldsB[2][64 * LDT];
  const int tid = threadIdx.x;
  const int lane = tid & 31;
  const int wave = tid >> 5;
  const int hlf = lane >> 4, l15 = lane & 15;
  const int n0 = blockIdx.x * 64;
  const int m0 = blockIdx.y * 64;
  const int nk = K >> 5;

  auto issue = [&](int ks, int b) {
    const int k0 = ks << 5;
#pragma unroll
    for (int i = 0; i < 4; ++i) {
      const int ch = tid + 128 * i;              // 0..511
      const int row = ch >> 3, c4 = (ch & 7) << 2;
      async_load16((unsigned)(size_t)(void*)&ldsA[b][row * LDT + c4],
                   A + (size_t)(m0 + row) * K + k0 + c4);
    }
#pragma unroll
    for (int i = 0; i < 4; ++i) {
      const int ch = tid + 128 * i;
      const int row = ch >> 3, c4 = (ch & 7) << 2;
      async_load16((unsigned)(size_t)(void*)&ldsB[b][row * LDT + c4],
                   W + (size_t)(n0 + row) * K + k0 + c4);
    }
  };

  v8f acc[4];
#pragma unroll
  for (int j = 0; j < 4; ++j) acc[j] = (v8f){};

  issue(0, 0);
  wait_async0();
  __syncthreads();

  for (int ks = 0; ks < nk; ++ks) {
    const int b = ks & 1;
    if (ks + 1 < nk) issue(ks + 1, b ^ 1);

    const float* Ab = &ldsA[b][(wave * 16 + l15) * LDT];
    v16h af;
#pragma unroll
    for (int v = 0; v < 8; ++v) {
      const int kb = (hlf << 4) + (v << 1);
      float2 p = *(const float2*)(Ab + kb);
      af[2 * v] = (f16)p.x;
      af[2 * v + 1] = (f16)p.y;
    }
#pragma unroll
    for (int j = 0; j < 4; ++j) {
      const float* Bb = &ldsB[b][(j * 16 + l15) * LDT + hlf * 16];
      v16h bf;
#pragma unroll
      for (int v = 0; v < 8; ++v) {
        float2 q = *(const float2*)(Bb + 2 * v);
        bf[2 * v] = (f16)q.x;
        bf[2 * v + 1] = (f16)q.y;
      }
      acc[j] = wmma16(af, bf, acc[j]);
    }
    wait_async0();
    __syncthreads();
  }

#pragma unroll
  for (int j = 0; j < 4; ++j) {
    const int n = n0 + j * 16 + l15;
    const float bv = bias ? bias[n] : 0.0f;
#pragma unroll
    for (int r = 0; r < 8; ++r) {
      const int m = m0 + wave * 16 + r + hlf * 8;
      Out[(size_t)m * N + n] = acc[j][r] + bv;
    }
  }
}

// ---------------------------------------------------------------- attention
__device__ __forceinline__ v16h load_kfrag(const float* __restrict__ KVb,
                                           int kbase, int kt16, int h, int lane) {
  const int hlf = lane >> 4, l15 = lane & 15;
  v16h f;
  const float* krow =
      KVb + (size_t)(kbase + kt16 * 16 + l15) * 576 + DSAC + h * DHEAD;
#pragma unroll
  for (int v = 0; v < 8; ++v) {
    const int d = (hlf << 4) + 2 * v;
    float x0 = 0.f, x1 = 0.f;
    if (d < DHEAD) { float2 p = *(const float2*)(krow + d); x0 = p.x; x1 = p.y; }
    f[2 * v] = (f16)x0; f[2 * v + 1] = (f16)x1;
  }
  return f;
}
__device__ __forceinline__ v16h load_vfrag(const float* __restrict__ KVb,
                                           int kbase, int k2, int Nk, int h,
                                           int lane, int dblock) {
  const int hlf = lane >> 4, l15 = lane & 15;
  const int dcol = dblock + l15;
  v16h f;
  if (dcol < DHEAD) {
    const float* col = KVb + 2 * DSAC + h * DHEAD + dcol;
#pragma unroll
    for (int v = 0; v < 8; ++v) {
      const int kk = (hlf << 4) + 2 * v;
      const int key = k2 + kk;
      float x0 = (key < Nk) ? col[(size_t)(kbase + key) * 576] : 0.f;
      float x1 = (key + 1 < Nk) ? col[(size_t)(kbase + key + 1) * 576] : 0.f;
      f[2 * v] = (f16)x0; f[2 * v + 1] = (f16)x1;
    }
  } else {
#pragma unroll
    for (int i = 0; i < 16; ++i) f[i] = (f16)0.f;
  }
  return f;
}

// Q rows [B*Nq,576] (q at h*24), KV rows [B*Nk,576] (k at 192+h*24, v at 384+h*24)
// Out rows [B*Nq,192]. Two-pass softmax; 16-query tile per wave.
__global__ __launch_bounds__(32) void k_attn(const float* __restrict__ Qb,
                                             const float* __restrict__ KVb,
                                             float* __restrict__ Out,
                                             int Nq, int Nk) {
  __shared__ _Float16 P[16 * 32];
  const int lane = threadIdx.x;
  const int hlf = lane >> 4, l15 = lane & 15;
  const int qtile = blockIdx.x;
  const int b = blockIdx.y >> 3, h = blockIdx.y & 7;
  const int qbase = b * Nq, kbase = b * Nk;

  v16h qa;
  {
    const float* qrow = Qb + (size_t)(qbase + qtile * 16 + l15) * 576 + h * DHEAD;
#pragma unroll
    for (int v = 0; v < 8; ++v) {
      const int kk = (hlf << 4) + 2 * v;
      float x0 = 0.f, x1 = 0.f;
      if (kk < DHEAD) { float2 p = *(const float2*)(qrow + kk); x0 = p.x; x1 = p.y; }
      qa[2 * v] = (f16)x0; qa[2 * v + 1] = (f16)x1;
    }
  }

  // pass 1: row max
  float rowmax[8];
#pragma unroll
  for (int r = 0; r < 8; ++r) rowmax[r] = -1e30f;
  const int ktiles = Nk >> 4;
  for (int kt = 0; kt < ktiles; ++kt) {
    v16h kf = load_kfrag(KVb, kbase, kt, h, lane);
    v8f z = {};
    v8f s = wmma16(qa, kf, z);
#pragma unroll
    for (int r = 0; r < 8; ++r)
      rowmax[r] = fmaxf(rowmax[r], redmax16(s[r] * ATT_SCALE));
  }

  // pass 2: exp/sum + P·V
  float rowsum[8] = {0, 0, 0, 0, 0, 0, 0, 0};
  v8f o0 = {}, o1 = {};
  for (int k2 = 0; k2 < Nk; k2 += 32) {
    {
      v16h kf = load_kfrag(KVb, kbase, k2 >> 4, h, lane);
      v8f z = {};
      v8f s = wmma16(qa, kf, z);
#pragma unroll
      for (int r = 0; r < 8; ++r) {
        float e = __expf(s[r] * ATT_SCALE - rowmax[r]);
        rowsum[r] += e;
        P[(r + hlf * 8) * 32 + l15] = (f16)e;
      }
    }
    if (k2 + 16 < Nk) {
      v16h kf = load_kfrag(KVb, kbase, (k2 >> 4) + 1, h, lane);
      v8f z = {};
      v8f s = wmma16(qa, kf, z);
#pragma unroll
      for (int r = 0; r < 8; ++r) {
        float e = __expf(s[r] * ATT_SCALE - rowmax[r]);
        rowsum[r] += e;
        P[(r + hlf * 8) * 32 + 16 + l15] = (f16)e;
      }
    } else {
#pragma unroll
      for (int r = 0; r < 8; ++r) P[(r + hlf * 8) * 32 + 16 + l15] = (f16)0.f;
    }
    __syncthreads();
    v16h pa;
#pragma unroll
    for (int v = 0; v < 8; ++v) {
      const int kk = (hlf << 4) + 2 * v;
      pa[2 * v] = P[l15 * 32 + kk];
      pa[2 * v + 1] = P[l15 * 32 + kk + 1];
    }
    v16h vf0 = load_vfrag(KVb, kbase, k2, Nk, h, lane, 0);
    v16h vf1 = load_vfrag(KVb, kbase, k2, Nk, h, lane, 16);
    o0 = wmma16(pa, vf0, o0);
    o1 = wmma16(pa, vf1, o1);
    __syncthreads();
  }

#pragma unroll
  for (int r = 0; r < 8; ++r) rowsum[r] = redsum16(rowsum[r]);
#pragma unroll
  for (int r = 0; r < 8; ++r) {
    const int m = qtile * 16 + r + hlf * 8;
    const float inv = 1.0f / rowsum[r];
    float* orow = Out + (size_t)(qbase + m) * DSAC + h * DHEAD;
    orow[l15] = o0[r] * inv;
    if (l15 < 8) orow[16 + l15] = o1[r] * inv;
  }
}

// ---------------------------------------------------------------- elementwise
__global__ void k_posconv(const float* __restrict__ x, const float* __restrict__ w9,
                          const float* __restrict__ bias, float* __restrict__ out) {
  int idx = blockIdx.x * blockDim.x + threadIdx.x;
  const int total = BATCH * DIMC * HWC;
  if (idx >= total) return;
  int xw = idx % WWD, yh = (idx / WWD) % HH, c = (idx / HWC) % DIMC,
      bb = idx / (HWC * DIMC);
  const float* xp = x + (size_t)(bb * DIMC + c) * HWC;
  const float* wc = w9 + c * 9;
  float acc = bias[c];
#pragma unroll
  for (int dy = -1; dy <= 1; ++dy)
#pragma unroll
    for (int dx = -1; dx <= 1; ++dx) {
      int yy = yh + dy, xx = xw + dx;
      if (yy >= 0 && yy < HH && xx >= 0 && xx < WWD)
        acc += xp[yy * WWD + xx] * wc[(dy + 1) * 3 + (dx + 1)];
    }
  out[idx] = x[idx] + acc;
}

__global__ void k_nchw2rows(const float* __restrict__ src, float* __restrict__ dst,
                            int C, int dstC, int dstOff,
                            const float* __restrict__ g, const float* __restrict__ bt,
                            const float* __restrict__ mn, const float* __restrict__ vr) {
  int idx = blockIdx.x * blockDim.x + threadIdx.x;
  const int total = BATCH * C * HWC;
  if (idx >= total) return;
  int hw = idx % HWC, c = (idx / HWC) % C, bb = idx / (HWC * C);
  float v = src[idx];
  if (g) {
    float inv = g[c] * rsqrtf(vr[c] + EPSB);
    v = v * inv + (bt[c] - mn[c] * inv);
  }
  dst[(size_t)(bb * HWC + hw) * dstC + dstOff + c] = v;
}

__global__ void k_dwconv(const float* __restrict__ Y1, const float* __restrict__ w9,
                         const float* __restrict__ bias, float* __restrict__ out,
                         const float* __restrict__ g, const float* __restrict__ bt,
                         const float* __restrict__ mn, const float* __restrict__ vr) {
  int idx = blockIdx.x * blockDim.x + threadIdx.x;
  const int total = BATCH * DCONVC * HWC;
  if (idx >= total) return;
  int xw = idx % WWD, yh = (idx / WWD) % HH, c = (idx / HWC) % DCONVC,
      bb = idx / (HWC * DCONVC);
  float inv = g[c] * rsqrtf(vr[c] + EPSB);
  float sh = bt[c] - mn[c] * inv;
  const float* base = Y1 + (size_t)bb * HWC * DIMC + DSAC + c;
  float center = base[(size_t)(yh * WWD + xw) * DIMC];
  const float* wc = w9 + c * 9;
  float acc = bias[c];
#pragma unroll
  for (int dy = -1; dy <= 1; ++dy)
#pragma unroll
    for (int dx = -1; dx <= 1; ++dx) {
      int yy = yh + dy, xx = xw + dx;
      if (yy >= 0 && yy < HH && xx >= 0 && xx < WWD) {
        float vv = base[(size_t)(yy * WWD + xx) * DIMC];
        acc += (vv * inv + sh) * wc[(dy + 1) * 3 + (dx + 1)];
      }
    }
  out[(size_t)(bb * DCONVC + c) * HWC + yh * WWD + xw] = center + acc;
}

__global__ void k_pool(const float* __restrict__ Y1, float* __restrict__ sa0) {
  int idx = blockIdx.x * blockDim.x + threadIdx.x;
  const int total = BATCH * N2C * DSAC;
  if (idx >= total) return;
  int c = idx % DSAC, n2 = (idx / DSAC) % N2C, bb = idx / (DSAC * N2C);
  int hd = n2 / WDS, wd = n2 % WDS, y = 2 * hd, x = 2 * wd;
  const float* base = Y1 + (size_t)bb * HWC * DIMC + c;
  float s = base[(size_t)(y * WWD + x) * DIMC] +
            base[(size_t)(y * WWD + x + 1) * DIMC] +
            base[(size_t)((y + 1) * WWD + x) * DIMC] +
            base[(size_t)((y + 1) * WWD + x + 1) * DIMC];
  sa0[idx] = 0.25f * s;
}

__global__ __launch_bounds__(32) void k_ln(const float* __restrict__ in,
                                           const float* __restrict__ g,
                                           const float* __restrict__ bt,
                                           float* __restrict__ out) {
  const int row = blockIdx.x, lane = threadIdx.x;
  const float* p = in + (size_t)row * DSAC;
  float v[6], s = 0.f;
#pragma unroll
  for (int j = 0; j < 6; ++j) { v[j] = p[lane + 32 * j]; s += v[j]; }
  s = redsum32(s);
  const float mu = s * (1.0f / DSAC);
  float q = 0.f;
#pragma unroll
  for (int j = 0; j < 6; ++j) { float d = v[j] - mu; q += d * d; }
  q = redsum32(q);
  const float inv = rsqrtf(q * (1.0f / DSAC) + EPSB);
#pragma unroll
  for (int j = 0; j < 6; ++j) {
    int c = lane + 32 * j;
    out[(size_t)row * DSAC + c] = (v[j] - mu) * inv * g[c] + bt[c];
  }
}

__global__ void k_add(const float* __restrict__ a, const float* __restrict__ b2,
                      float* __restrict__ out, int n) {
  int idx = blockIdx.x * blockDim.x + threadIdx.x;
  if (idx < n) out[idx] = a[idx] + b2[idx];
}

__global__ void k_rows2nchw_add(const float* __restrict__ rows,
                                const float* __restrict__ res,
                                float* __restrict__ out, int C, int ld, int off) {
  int idx = blockIdx.x * blockDim.x + threadIdx.x;
  const int total = BATCH * C * HWC;
  if (idx >= total) return;
  int hw = idx % HWC, c = (idx / HWC) % C, bb = idx / (HWC * C);
  float v = rows[(size_t)(bb * HWC + hw) * ld + off + c];
  out[idx] = (res ? res[idx] : 0.f) + v;
}

__global__ void k_gelu(float* __restrict__ x, int n) {
  int idx = blockIdx.x * blockDim.x + threadIdx.x;
  if (idx < n) {
    float v = x[idx];
    x[idx] = 0.5f * v * (1.0f + erff(v * 0.7071067811865475f));
  }
}

__global__ void k_upsample(const float* __restrict__ saf, float* __restrict__ out) {
  int idx = blockIdx.x * blockDim.x + threadIdx.x;
  const int total = BATCH * DSAC * HWC;
  if (idx >= total) return;
  int xw = idx % WWD, yh = (idx / WWD) % HH, c = (idx / HWC) % DSAC,
      bb = idx / (HWC * DSAC);
  float fy = 0.5f * yh - 0.25f, fx = 0.5f * xw - 0.25f;
  int y0 = (int)floorf(fy), x0 = (int)floorf(fx);
  float wy = fy - y0, wx = fx - x0;
  int y1 = y0 + 1, x1 = x0 + 1;
  y0 = min(max(y0, 0), HDS - 1); y1 = min(max(y1, 0), HDS - 1);
  x0 = min(max(x0, 0), WDS - 1); x1 = min(max(x1, 0), WDS - 1);
  const float* base = saf + (size_t)bb * N2C * DSAC + c;
  float v00 = base[(size_t)(y0 * WDS + x0) * DSAC];
  float v01 = base[(size_t)(y0 * WDS + x1) * DSAC];
  float v10 = base[(size_t)(y1 * WDS + x0) * DSAC];
  float v11 = base[(size_t)(y1 * WDS + x1) * DSAC];
  out[idx] = (1.f - wy) * ((1.f - wx) * v00 + wx * v01) +
             wy * ((1.f - wx) * v10 + wx * v11);
}

// ---------------------------------------------------------------- launcher
extern "C" void kernel_launch(void* const* d_in, const int* in_sizes, int n_in,
                              void* d_out, int out_size, void* d_ws, size_t ws_size,
                              hipStream_t stream) {
  const float* x      = (const float*)d_in[0];
  const float* pos_w  = (const float*)d_in[1];
  const float* pos_b  = (const float*)d_in[2];
  const float* bn1_g  = (const float*)d_in[3];
  const float* bn1_b  = (const float*)d_in[4];
  const float* bn1_m  = (const float*)d_in[5];
  const float* bn1_v  = (const float*)d_in[6];
  const float* conv1_w= (const float*)d_in[7];
  const float* conv1_b= (const float*)d_in[8];
  const float* bnc1_g = (const float*)d_in[9];
  const float* bnc1_b = (const float*)d_in[10];
  const float* bnc1_m = (const float*)d_in[11];
  const float* bnc1_v = (const float*)d_in[12];
  const float* dw_w   = (const float*)d_in[13];
  const float* dw_b   = (const float*)d_in[14];
  const float* lns1_g = (const float*)d_in[15];
  const float* lns1_b = (const float*)d_in[16];
  const float* chup_w = (const float*)d_in[17];
  const float* chup_b = (const float*)d_in[18];
  const float* bnc2_g = (const float*)d_in[19];
  const float* bnc2_b = (const float*)d_in[20];
  const float* bnc2_m = (const float*)d_in[21];
  const float* bnc2_v = (const float*)d_in[22];
  const float* lns2_g = (const float*)d_in[23];
  const float* lns2_b = (const float*)d_in[24];
  const float* ccuc_w = (const float*)d_in[25];
  const float* ccuc_b = (const float*)d_in[26];
  const float* ccus_w = (const float*)d_in[27];
  const float* ccus_b = (const float*)d_in[28];
  const float* fc_w   = (const float*)d_in[29];
  const float* fc_b   = (const float*)d_in[30];
  const float* fs_w   = (const float*)d_in[31];
  const float* fs_b   = (const float*)d_in[32];
  const float* conv2_w= (const float*)d_in[33];
  const float* conv2_b= (const float*)d_in[34];
  const float* bn2_g  = (const float*)d_in[35];
  const float* bn2_b  = (const float*)d_in[36];
  const float* bn2_m  = (const float*)d_in[37];
  const float* bn2_v  = (const float*)d_in[38];
  const float* fc1_w  = (const float*)d_in[39];
  const float* fc1_b  = (const float*)d_in[40];
  const float* fc2_w  = (const float*)d_in[41];
  const float* fc2_b  = (const float*)d_in[42];

  const size_t S_nchw384 = (size_t)BATCH * DIMC * HWC;    // 4,816,896
  const size_t S_big     = (size_t)BATCH * HWC * 576;     // 7,225,344
  const size_t S_nchw192 = (size_t)BATCH * DCONVC * HWC;  // 2,408,448
  const size_t S_r192s   = (size_t)BATCH * N2C * DSAC;    //   602,112
  const size_t S_r576s   = (size_t)BATCH * N2C * 576;     // 1,806,336
  const size_t S_hid     = (size_t)BATCH * HWC * HIDC;    // 19,267,584

  float* ws = (float*)d_ws;
  size_t off = 0;
  auto alloc = [&](size_t n) { float* p = ws + off; off += n; return p; };
  float* x1     = alloc(S_nchw384);  // x after pos conv; later x2 (in place)
  float* Arows  = alloc(S_nchw384);  // shared A-matrix staging
  float* big    = alloc(S_big);      // Y1 rows, then conv_qkv rows
  float* c24a   = alloc(S_nchw192);  // conv_res NCHW, then sa_up NCHW
  float* sa0    = alloc(S_r192s);
  float* sbuf   = alloc(S_r576s);    // chup out, then sa_qkv
  float* small1 = alloc(S_r192s);    // O1, then O3, then saf
  float* sa1    = alloc(S_r192s);
  float* small2 = alloc(S_r192s);    // fs out
  float* c24b   = alloc(S_nchw192);  // O2 rows, then conv_final NCHW
  float* c24c   = alloc(S_nchw192);  // fc out rows
  float* big2   = alloc(S_nchw384);  // Y2 rows, then fc2 out rows
  float* hid    = alloc(S_hid);      // fc1 out rows

  const int T = 256;
  auto cd = [](size_t a, int b) { return (unsigned)((a + b - 1) / b); };
  const int Mbig = BATCH * HWC;  // 12544
  const int Msa  = BATCH * N2C;  //  3136

  // 1. x1 = x + dwconv3x3(x, pos)
  k_posconv<<<cd(S_nchw384, T), T, 0, stream>>>(x, pos_w, pos_b, x1);
  // 2. A1 = rows(bn1(x1))
  k_nchw2rows<<<cd(S_nchw384, T), T, 0, stream>>>(x1, Arows, DIMC, DIMC, 0,
                                                  bn1_g, bn1_b, bn1_m, bn1_v);
  // 3. Y1 = A1 @ conv1_w^T + b        [12544 x 384]
  k_gemm<<<dim3(DIMC / 64, Mbig / 64), 128, 0, stream>>>(
      Arows, conv1_w, conv1_b, big, Mbig, DIMC, DIMC);
  // 4. conv_res = conv + dwconv(bnc1(conv))   (NCHW)
  k_dwconv<<<cd(S_nchw192, T), T, 0, stream>>>(big, dw_w, dw_b, c24a,
                                               bnc1_g, bnc1_b, bnc1_m, bnc1_v);
  // 5. sa0 = 2x2 avg pool of qkv channels (exact bilinear 1/2x)
  k_pool<<<cd(S_r192s, T), T, 0, stream>>>(big, sa0);
  // 6. LN1
  k_ln<<<Msa, 32, 0, stream>>>(sa0, lns1_g, lns1_b, Arows);
  // 7. s = LN1 @ chup_w^T + b         [3136 x 576]
  k_gemm<<<dim3(576 / 64, Msa / 64), 128, 0, stream>>>(
      Arows, chup_w, chup_b, sbuf, Msa, 576, DSAC);
  // 8. attn1 (self)
  k_attn<<<dim3(N2C / 16, BATCH * HEADS), 32, 0, stream>>>(sbuf, sbuf, small1,
                                                           N2C, N2C);
  // 9. sa1 = sa0 + attn1
  k_add<<<cd(S_r192s, T), T, 0, stream>>>(sa0, small1, sa1, (int)S_r192s);
  // 10. LN2
  k_ln<<<Msa, 32, 0, stream>>>(sa1, lns2_g, lns2_b, Arows);
  // 11. sa_qkv = LN2 @ ccus_w^T + b   [3136 x 576]  (reuses sbuf)
  k_gemm<<<dim3(576 / 64, Msa / 64), 128, 0, stream>>>(
      Arows, ccus_w, ccus_b, sbuf, Msa, 576, DSAC);
  // 12. rows(bnc2(conv_res))
  k_nchw2rows<<<cd(S_nchw192, T), T, 0, stream>>>(c24a, Arows, DCONVC, DCONVC, 0,
                                                  bnc2_g, bnc2_b, bnc2_m, bnc2_v);
  // 13. conv_qkv = @ ccuc_w^T + b     [12544 x 576] (reuses big)
  k_gemm<<<dim3(576 / 64, Mbig / 64), 128, 0, stream>>>(
      Arows, ccuc_w, ccuc_b, big, Mbig, 576, DCONVC);
  // 14. O2 = attn(cq | sk,sv)
  k_attn<<<dim3(HWC / 16, BATCH * HEADS), 32, 0, stream>>>(big, sbuf, c24b,
                                                           HWC, N2C);
  // 15. O3 = attn(sq | ck,cv)
  k_attn<<<dim3(N2C / 16, BATCH * HEADS), 32, 0, stream>>>(sbuf, big, small1,
                                                           N2C, HWC);
  // 16. fc: O2 @ fc_w^T + b
  k_gemm<<<dim3(DCONVC / 64, Mbig / 64), 128, 0, stream>>>(
      c24b, fc_w, fc_b, c24c, Mbig, DCONVC, DCONVC);
  // 17. conv_final = conv_res + transpose(fc out)   (NCHW, into c24b)
  k_rows2nchw_add<<<cd(S_nchw192, T), T, 0, stream>>>(c24c, c24a, c24b, DCONVC,
                                                      DCONVC, 0);
  // 18. fs: O3 @ fs_w^T + b
  k_gemm<<<dim3(DSAC / 64, Msa / 64), 128, 0, stream>>>(
      small1, fs_w, fs_b, small2, Msa, DSAC, DSAC);
  // 19. saf = sa1 + fs out
  k_add<<<cd(S_r192s, T), T, 0, stream>>>(sa1, small2, small1, (int)S_r192s);
  // 20. sa_up = bilinear 2x upsample(saf)  (NCHW, into c24a)
  k_upsample<<<cd(S_nchw192, T), T, 0, stream>>>(small1, c24a);
  // 21. concat rows: [conv_final | sa_up] -> [12544 x 384]
  k_nchw2rows<<<cd(S_nchw192, T), T, 0, stream>>>(c24b, Arows, DCONVC, DIMC, 0,
                                                  nullptr, nullptr, nullptr, nullptr);
  k_nchw2rows<<<cd(S_nchw192, T), T, 0, stream>>>(c24a, Arows, DSAC, DIMC, DCONVC,
                                                  nullptr, nullptr, nullptr, nullptr);
  // 22. Y2 = concat @ conv2_w^T + b
  k_gemm<<<dim3(DIMC / 64, Mbig / 64), 128, 0, stream>>>(
      Arows, conv2_w, conv2_b, big2, Mbig, DIMC, DIMC);
  // 23. x2 = x1 + Y2  (in place over x1)
  k_rows2nchw_add<<<cd(S_nchw384, T), T, 0, stream>>>(big2, x1, x1, DIMC, DIMC, 0);
  // 24. rows(bn2(x2))
  k_nchw2rows<<<cd(S_nchw384, T), T, 0, stream>>>(x1, Arows, DIMC, DIMC, 0,
                                                  bn2_g, bn2_b, bn2_m, bn2_v);
  // 25. hid = @ fc1_w^T + b           [12544 x 1536]
  k_gemm<<<dim3(HIDC / 64, Mbig / 64), 128, 0, stream>>>(
      Arows, fc1_w, fc1_b, hid, Mbig, HIDC, DIMC);
  // 26. exact GeLU in place
  k_gelu<<<cd(S_hid, T), T, 0, stream>>>(hid, (int)S_hid);
  // 27. y = gelu @ fc2_w^T + b        [12544 x 384]
  k_gemm<<<dim3(DIMC / 64, Mbig / 64), 128, 0, stream>>>(
      hid, fc2_w, fc2_b, big2, Mbig, DIMC, HIDC);
  // 28. out = x2 + y   (NCHW, f32)
  k_rows2nchw_add<<<cd(S_nchw384, T), T, 0, stream>>>(big2, x1, (float*)d_out,
                                                      DIMC, DIMC, 0);
}
